// Temporal_Learning_14010183320065
// MI455X (gfx1250) — matrix operand order repaired and checked by reading, not compiled, not run on hardware
//
#include <hip/hip_runtime.h>

#define B_   64
#define S_   2048
#define I_   3
#define H_   256
#define T_   (S_ - I_)
#define NWG  16
#define NTH  256

typedef _Float16 v16h  __attribute__((ext_vector_type(16)));
typedef _Float16 v8h   __attribute__((ext_vector_type(8)));
typedef float    v8f   __attribute__((ext_vector_type(8)));
typedef unsigned int u32x4 __attribute__((ext_vector_type(4)));
typedef int          i32x8 __attribute__((ext_vector_type(8)));
typedef int          i32x4 __attribute__((ext_vector_type(4)));

#if __has_builtin(__builtin_amdgcn_tensor_load_to_lds)
#define HAVE_TDM 1
#else
#define HAVE_TDM 0
#endif

// ---- LDS layout (bytes) ----
#define OFF_WH1   0         // 64x256 f16  W_hh1 slice (rows: i,f,g,o x 16)
#define OFF_WI2   32768     // 64x256 f16  W_ih2 slice
#define OFF_WH2   65536     // 64x256 f16  W_hh2 slice
#define OFF_HB    98304     // 64x256 f16  staged h2_prev (TDM target)
#define OFF_GATES 131072    // 64x64 f32   gate accumulators
#define OFF_C1    147456    // 64x16 f32   c1 slice
#define OFF_C2    151552    // 64x16 f32   c2 slice
#define OFF_WIH1  155648    // 64x3  f32   W_ih1 slice
#define OFF_B1    156416    // 64 f32      b_ih1+b_hh1 slice
#define OFF_B2    156672    // 64 f32      b_ih2+b_hh2 slice
#define OFF_WOUT  156928    // 256 f32     W_out
#define SMEM_SZ   157952

__device__ __forceinline__ float sigm_(float v) {
  return 1.0f / (1.0f + __expf(-v));
}
__device__ __forceinline__ float tanh_(float v) {
  v = fminf(fmaxf(v, -15.0f), 15.0f);
  float e = __expf(2.0f * v);
  return (e - 1.0f) / (e + 1.0f);
}

// A fragment (16x32 f16): row-major source, ld = H_.
// lanes 0-15: row M=lane, K = kb*32 + {0..7, 16..23}
// lanes16-31: row M=lane-16, K = kb*32 + {8..15, 24..31}
__device__ __forceinline__ v16h loadA(const _Float16* base, int row0, int kb, int lane) {
  int row = row0 + (lane & 15);
  int kh  = lane >> 4;
  const _Float16* p = base + row * H_ + kb * 32 + kh * 8;
  v8h lo = *(const v8h*)p;
  v8h hi = *(const v8h*)(p + 16);
  v16h a;
#pragma unroll
  for (int i = 0; i < 8; ++i) { a[i] = lo[i]; a[i + 8] = hi[i]; }
  return a;
}

// B fragment (32x16 f16): weights stored n-major (n x 256), B[k][n] = W[n][k].
// lane L<16: col = L, K = kb*32 + {0..7, 16..23}; L>=16: K = kb*32 + {8..15, 24..31}.
__device__ __forceinline__ v16h loadB(const _Float16* w, int col0, int kb, int lane) {
  int n  = col0 + (lane & 15);
  int kh = lane >> 4;
  const _Float16* p = w + n * H_ + kb * 32 + kh * 8;
  v8h lo = *(const v8h*)p;
  v8h hi = *(const v8h*)(p + 16);
  v16h b;
#pragma unroll
  for (int i = 0; i < 8; ++i) { b[i] = lo[i]; b[i + 8] = hi[i]; }
  return b;
}

__device__ __forceinline__ v8f wmma_(v16h a, v16h b, v8f c) {
  return __builtin_amdgcn_wmma_f32_16x16x32_f16(false, a, false, b, (short)0, c, false, false);
}

#if HAVE_TDM
// Issue a TDM copy of 32768 bytes (64x256 f16) global -> LDS[lds_off].
// D# per ISA 8.3/8.4: group0 {count=1, lds_addr, global_addr, type=2};
// group1 {data_size=3 (8B), tensor_dim0=4096, tile_dim0=4096, stride=4096}.
// 6-arg builtin (clang-23 / therock): (g0, g1, g2, g3, g_extra, cpol).
__device__ __forceinline__ void tdm_load_h(const void* gptr, unsigned lds_off) {
  unsigned long long ga = (unsigned long long)gptr;
  u32x4 g0;
  g0[0] = 1u;                                        // count=1, user mode
  g0[1] = lds_off;                                   // lds_addr
  g0[2] = (unsigned)(ga & 0xFFFFFFFFu);              // global_addr[31:0]
  g0[3] = (unsigned)((ga >> 32) & 0x1FFFFFFu) | (2u << 30);  // addr[56:32] | type=2
  i32x8 g1;
  g1[0] = (int)(3u << 16);                           // mask=0 | data_size=8B
  g1[1] = (int)(4096u << 16);                        // tensor_dim0[15:0] << 16
  g1[2] = 0;                                         // tensor_dim0[31:16] | tensor_dim1 lo
  g1[3] = (int)(4096u << 16);                        // tile_dim0 = 4096
  g1[4] = 0;                                         // tile_dim1/2 unused
  g1[5] = 4096;                                      // tensor_dim0_stride[31:0]
  g1[6] = 0;
  g1[7] = 0;
  i32x4 z4 = {0, 0, 0, 0};
  i32x8 z8 = {0, 0, 0, 0, 0, 0, 0, 0};
  __builtin_amdgcn_tensor_load_to_lds(g0, g1, z4, z4, z8, 0);
}
#endif

// Grid-wide sense-reversing barrier over NWG workgroups.
__device__ __forceinline__ void grid_sync(unsigned* bar) {
  __threadfence();            // release this thread's global stores
  __syncthreads();
  if (threadIdx.x == 0) {
    unsigned g = __hip_atomic_load(&bar[1], __ATOMIC_RELAXED, __HIP_MEMORY_SCOPE_AGENT);
    unsigned prev = __hip_atomic_fetch_add(&bar[0], 1u, __ATOMIC_ACQ_REL, __HIP_MEMORY_SCOPE_AGENT);
    if (prev == NWG - 1u) {
      __hip_atomic_store(&bar[0], 0u, __ATOMIC_RELAXED, __HIP_MEMORY_SCOPE_AGENT);
      __hip_atomic_fetch_add(&bar[1], 1u, __ATOMIC_RELEASE, __HIP_MEMORY_SCOPE_AGENT);
    } else {
      while (__hip_atomic_load(&bar[1], __ATOMIC_RELAXED, __HIP_MEMORY_SCOPE_AGENT) == g)
        __builtin_amdgcn_s_sleep(2);
    }
    __builtin_amdgcn_fence(__ATOMIC_ACQUIRE, "agent");  // invalidate WGP$ for fresh h state
  }
  __syncthreads();
}

extern "C" __global__ void __launch_bounds__(NTH, 1)
lstm_persistent(const float* __restrict__ x,
                const float* __restrict__ W_ih1, const float* __restrict__ b_ih1,
                const float* __restrict__ W_hh1, const float* __restrict__ b_hh1,
                const float* __restrict__ W_ih2, const float* __restrict__ b_ih2,
                const float* __restrict__ W_hh2, const float* __restrict__ b_hh2,
                const float* __restrict__ W_out, const float* __restrict__ b_out,
                float* __restrict__ out,
                _Float16* __restrict__ H1, _Float16* __restrict__ H2,
                unsigned* __restrict__ bar)
{
  __shared__ __align__(16) unsigned char smem[SMEM_SZ];
  _Float16* Wh1s  = (_Float16*)(smem + OFF_WH1);
  _Float16* Wi2s  = (_Float16*)(smem + OFF_WI2);
  _Float16* Wh2s  = (_Float16*)(smem + OFF_WH2);
  _Float16* hB    = (_Float16*)(smem + OFF_HB);
  float*    gatesS= (float*)(smem + OFF_GATES);
  float*    c1s   = (float*)(smem + OFF_C1);
  float*    c2s   = (float*)(smem + OFF_C2);
  float*    Wih1L = (float*)(smem + OFF_WIH1);
  float*    b1L   = (float*)(smem + OFF_B1);
  float*    b2L   = (float*)(smem + OFF_B2);
  float*    WoutL = (float*)(smem + OFF_WOUT);

  const int tid  = threadIdx.x;
  const int lane = tid & 31;
  const int wave = tid >> 5;
  const int wg   = blockIdx.x;
  const int hs   = wg * 16;   // this WG owns hidden units [hs, hs+16)

  // ---- prologue: resident weights (f32 -> f16), biases, zero cell state ----
  for (int idx = tid; idx < 64 * H_; idx += NTH) {
    int lr = idx >> 8;             // local gate row (q*16+j)
    int k  = idx & 255;
    int gr = (lr >> 4) * H_ + hs + (lr & 15);   // global gate row
    Wh1s[idx] = (_Float16)W_hh1[gr * H_ + k];
    Wi2s[idx] = (_Float16)W_ih2[gr * H_ + k];
    Wh2s[idx] = (_Float16)W_hh2[gr * H_ + k];
  }
  for (int lr = tid; lr < 64; lr += NTH) {
    int gr = (lr >> 4) * H_ + hs + (lr & 15);
    Wih1L[lr * 3 + 0] = W_ih1[gr * I_ + 0];
    Wih1L[lr * 3 + 1] = W_ih1[gr * I_ + 1];
    Wih1L[lr * 3 + 2] = W_ih1[gr * I_ + 2];
    b1L[lr] = b_ih1[gr] + b_hh1[gr];
    b2L[lr] = b_ih2[gr] + b_hh2[gr];
  }
  for (int k = tid; k < H_; k += NTH) WoutL[k] = W_out[k];
  for (int idx = tid; idx < 64 * 16; idx += NTH) { c1s[idx] = 0.0f; c2s[idx] = 0.0f; }
  __syncthreads();

  const float bout0 = b_out[0];
  const int mt  = wave & 3;            // batch tile (rows mt*16..)
  const int nt0 = (wave >> 2) * 2;     // gate-col tiles
  const int nt1 = nt0 + 1;

  for (int t = 0; t < T_; ++t) {
    const int p = t & 1;
    const _Float16* h1cur = H1 + p * (B_ * H_);
    _Float16*       h1nxt = H1 + (1 - p) * (B_ * H_);
    const _Float16* h2cur = H2 + p * (B_ * H_);
    _Float16*       h2nxt = H2 + (1 - p) * (B_ * H_);

    // stage h2_prev into LDS (overlaps with phase-A GEMM)
#if HAVE_TDM
    if (wave == 0) tdm_load_h((const void*)h2cur, (unsigned)OFF_HB);
#else
    for (int idx = tid; idx < (B_ * H_ * 2) / 16; idx += NTH)
      ((uint4*)hB)[idx] = ((const uint4*)h2cur)[idx];
#endif

    // ---- phase A: gates1 slice = h1 @ Whh1_slice^T ----
    v8f a0 = {0.f,0.f,0.f,0.f,0.f,0.f,0.f,0.f};
    v8f a1 = {0.f,0.f,0.f,0.f,0.f,0.f,0.f,0.f};
#pragma unroll
    for (int kb = 0; kb < 8; ++kb) {
      v16h A = loadA(h1cur, mt * 16, kb, lane);
      a0 = wmma_(A, loadB(Wh1s, nt0 * 16, kb, lane), a0);
      a1 = wmma_(A, loadB(Wh1s, nt1 * 16, kb, lane), a1);
    }
    {
      int rbase = mt * 16 + ((lane >> 4) << 3);
      int col   = lane & 15;
#pragma unroll
      for (int r = 0; r < 8; ++r) {
        gatesS[(rbase + r) * 64 + nt0 * 16 + col] = a0[r];
        gatesS[(rbase + r) * 64 + nt1 * 16 + col] = a1[r];
      }
    }
    __syncthreads();

    // layer-1 cell: fuse xproj (I=3) + bias, activations, write h1_new slice
    {
      int j = tid & 15;
      int brow = tid >> 4;
#pragma unroll
      for (int bb = 0; bb < 4; ++bb) {
        int b = brow + bb * 16;
        float x0 = x[b * S_ + t], x1v = x[b * S_ + t + 1], x2v = x[b * S_ + t + 2];
        float gq[4];
#pragma unroll
        for (int q = 0; q < 4; ++q) {
          int gc = q * 16 + j;
          gq[q] = gatesS[b * 64 + gc] + x0 * Wih1L[gc * 3] + x1v * Wih1L[gc * 3 + 1]
                + x2v * Wih1L[gc * 3 + 2] + b1L[gc];
        }
        float iv = sigm_(gq[0]), fv = sigm_(gq[1]), gv = tanh_(gq[2]), ov = sigm_(gq[3]);
        float c  = c1s[b * 16 + j];
        float cn = fv * c + iv * gv;
        c1s[b * 16 + j] = cn;
        h1nxt[b * H_ + hs + j] = (_Float16)(ov * tanh_(cn));
      }
    }
    grid_sync(bar);   // h1_new visible to all WGs

#if HAVE_TDM
    if (wave == 0) {
#if __has_builtin(__builtin_amdgcn_s_wait_tensorcnt)
      __builtin_amdgcn_s_wait_tensorcnt(0);
#else
      asm volatile("s_wait_tensorcnt 0" ::: "memory");
#endif
    }
    __syncthreads();
#endif

    // ---- phase B: gates2 slice = h1_new @ Wih2^T + h2_prev @ Whh2^T ----
    v8f d0 = {0.f,0.f,0.f,0.f,0.f,0.f,0.f,0.f};
    v8f d1 = {0.f,0.f,0.f,0.f,0.f,0.f,0.f,0.f};
#pragma unroll
    for (int kb = 0; kb < 8; ++kb) {
      v16h A1 = loadA((const _Float16*)h1nxt, mt * 16, kb, lane);  // global
      v16h A2 = loadA(hB, mt * 16, kb, lane);                       // LDS (TDM-staged)
      d0 = wmma_(A1, loadB(Wi2s, nt0 * 16, kb, lane), d0);
      d0 = wmma_(A2, loadB(Wh2s, nt0 * 16, kb, lane), d0);
      d1 = wmma_(A1, loadB(Wi2s, nt1 * 16, kb, lane), d1);
      d1 = wmma_(A2, loadB(Wh2s, nt1 * 16, kb, lane), d1);
    }
    {
      int rbase = mt * 16 + ((lane >> 4) << 3);
      int col   = lane & 15;
#pragma unroll
      for (int r = 0; r < 8; ++r) {
        gatesS[(rbase + r) * 64 + nt0 * 16 + col] = d0[r];
        gatesS[(rbase + r) * 64 + nt1 * 16 + col] = d1[r];
      }
    }
    __syncthreads();

    // WG0: output column for step t-1 from staged h2_prev
    if (wg == 0 && t > 0) {
      int b = tid >> 2, part = tid & 3;
      const _Float16* hp = hB + b * H_ + part * 64;
      const float*    wp = WoutL + part * 64;
      float s = 0.0f;
#pragma unroll 8
      for (int j2 = 0; j2 < 64; ++j2) s += (float)hp[j2] * wp[j2];
      s += __shfl_xor(s, 1);
      s += __shfl_xor(s, 2);
      if (part == 0) out[b * T_ + (t - 1)] = s + bout0;
    }

    // layer-2 cell
    {
      int j = tid & 15;
      int brow = tid >> 4;
#pragma unroll
      for (int bb = 0; bb < 4; ++bb) {
        int b = brow + bb * 16;
        float gq[4];
#pragma unroll
        for (int q = 0; q < 4; ++q) {
          int gc = q * 16 + j;
          gq[q] = gatesS[b * 64 + gc] + b2L[gc];
        }
        float iv = sigm_(gq[0]), fv = sigm_(gq[1]), gv = tanh_(gq[2]), ov = sigm_(gq[3]);
        float c  = c2s[b * 16 + j];
        float cn = fv * c + iv * gv;
        c2s[b * 16 + j] = cn;
        h2nxt[b * H_ + hs + j] = (_Float16)(ov * tanh_(cn));
      }
    }
    grid_sync(bar);   // h2_new visible to all WGs
  }

  // final output column from last h2 state
  if (wg == 0) {
    const _Float16* hf = H2 + (T_ & 1) * (B_ * H_);
    int b = tid >> 2, part = tid & 3;
    const _Float16* hp = hf + b * H_ + part * 64;
    const float*    wp = WoutL + part * 64;
    float s = 0.0f;
#pragma unroll 8
    for (int j2 = 0; j2 < 64; ++j2) s += (float)hp[j2] * wp[j2];
    s += __shfl_xor(s, 1);
    s += __shfl_xor(s, 2);
    if (part == 0) out[b * T_ + (T_ - 1)] = s + bout0;
  }
}

extern "C" __global__ void init_ws(unsigned* ws, int n) {
  int i = blockIdx.x * blockDim.x + threadIdx.x;
  if (i < n) ws[i] = 0u;
}

extern "C" void kernel_launch(void* const* d_in, const int* in_sizes, int n_in,
                              void* d_out, int out_size, void* d_ws, size_t ws_size,
                              hipStream_t stream) {
  const float* x     = (const float*)d_in[0];
  const float* W_ih1 = (const float*)d_in[1];
  const float* b_ih1 = (const float*)d_in[2];
  const float* W_hh1 = (const float*)d_in[3];
  const float* b_hh1 = (const float*)d_in[4];
  const float* W_ih2 = (const float*)d_in[5];
  const float* b_ih2 = (const float*)d_in[6];
  const float* W_hh2 = (const float*)d_in[7];
  const float* b_hh2 = (const float*)d_in[8];
  const float* W_out = (const float*)d_in[9];
  const float* b_out = (const float*)d_in[10];
  float* out = (float*)d_out;

  unsigned char* ws = (unsigned char*)d_ws;
  _Float16* H1  = (_Float16*)(ws);                 // 2 x 64x256 f16 = 65536 B
  _Float16* H2  = (_Float16*)(ws + 65536);         // 2 x 64x256 f16 = 65536 B
  unsigned* bar = (unsigned*)(ws + 131072);        // {count, generation}

  const int nz = (131072 + 8) / 4;                 // zero h buffers + barrier
  init_ws<<<(nz + 255) / 256, 256, 0, stream>>>((unsigned*)d_ws, nz);
  lstm_persistent<<<NWG, NTH, 0, stream>>>(x, W_ih1, b_ih1, W_hh1, b_hh1,
                                           W_ih2, b_ih2, W_hh2, b_hh2,
                                           W_out, b_out, out, H1, H2, bar);
}
